// FluidAttentionLayer_67903432950551
// MI455X (gfx1250) — matrix operand order repaired and checked by reading, not compiled, not run on hardware
//
#include <hip/hip_runtime.h>

typedef __bf16 bf16_t;
typedef bf16_t v16bf __attribute__((ext_vector_type(16)));
typedef bf16_t v8bf  __attribute__((ext_vector_type(8)));
typedef bf16_t v4bf  __attribute__((ext_vector_type(4)));
typedef float  v8f   __attribute__((ext_vector_type(8)));
typedef float  v4f   __attribute__((ext_vector_type(4)));

#define S_LEN 4096
#define HID   2048
#define NH    16
#define NKV   4
#define DHEAD 128
#define QPG   4
#define GROUP ((QPG + 2) * DHEAD)   // 768
#define TOTAL (NKV * GROUP)         // 3072

// ---------- helpers ----------
__device__ __forceinline__ bf16_t f2bf(float f) {
    unsigned u = __builtin_bit_cast(unsigned, f);
    u += 0x7FFFu + ((u >> 16) & 1u);                 // round-to-nearest-even
    unsigned short h = (unsigned short)(u >> 16);
    return __builtin_bit_cast(bf16_t, h);
}

__device__ __forceinline__ v16bf cat8(v8bf lo, v8bf hi) {
    return __builtin_shufflevector(lo, hi, 0,1,2,3,4,5,6,7,8,9,10,11,12,13,14,15);
}

__device__ __forceinline__ v8f zero8() {
    v8f z = {0.f,0.f,0.f,0.f,0.f,0.f,0.f,0.f};
    return z;
}

// CDNA5 async global->LDS copy: per-lane 16B, tracked by ASYNCcnt.
typedef __attribute__((address_space(3))) void lds_void;
__device__ __forceinline__ void async_g2l_b128(void* lds, const void* g) {
    unsigned loff = (unsigned)(__SIZE_TYPE__)(lds_void*)lds;
    asm volatile("global_load_async_to_lds_b128 %0, %1, off"
                 :: "v"(loff), "v"(g) : "memory");
}
__device__ __forceinline__ void wait_async0() {
    asm volatile("s_wait_asynccnt 0" ::: "memory");
}

// ================================================================
// GEMM: C[M,N] = A[M,K] * W[N,K]^T
// TA = float (convert-stage) or bf16 (async-stage); W fp32; TO = float/bf16.
// 256 threads = 8 waves; block tile 128x128; K-step 64; ping-pong LDS.
// ================================================================
#define LDT 72   // LDS row stride in halves (144B, keeps b128 reads aligned)

template<int N, int K, typename TA, typename TO>
__global__ __launch_bounds__(256)
void gemm_xwT(const TA* __restrict__ A, const float* __restrict__ W,
              TO* __restrict__ C) {
    __shared__ bf16_t as[2][128 * LDT];
    __shared__ bf16_t bs[2][128 * LDT];

    const int tid  = threadIdx.x;
    const int lane = tid & 31;
    const int w    = tid >> 5;       // 0..7
    const int wm   = w & 3;          // 4 M-groups of 32 rows
    const int wn   = w >> 2;         // 2 N-groups of 64 cols
    const int m0   = blockIdx.y * 128;
    const int n0   = blockIdx.x * 128;
    const int lrow = lane & 15;
    const int lhi  = lane >> 4;

    v8f acc[2][4];
    #pragma unroll
    for (int i = 0; i < 2; ++i)
        #pragma unroll
        for (int j = 0; j < 4; ++j)
            acc[i][j] = zero8();

    auto stageW = [&](bf16_t* dst, int k0) {
        #pragma unroll
        for (int i = 0; i < 8; ++i) {
            int c   = tid + i * 256;       // 2048 float4 chunks
            int row = c >> 4;
            int col = (c & 15) * 4;
            v4f wv = *(const v4f*)(W + (size_t)(n0 + row) * K + k0 + col);
            v4bf wb = { f2bf(wv.x), f2bf(wv.y), f2bf(wv.z), f2bf(wv.w) };
            *(v4bf*)(dst + row * LDT + col) = wb;
        }
    };
    auto stageA = [&](bf16_t* dst, int k0) {
        if constexpr (sizeof(TA) == 4) {
            #pragma unroll
            for (int i = 0; i < 8; ++i) {
                int c   = tid + i * 256;
                int row = c >> 4;
                int col = (c & 15) * 4;
                v4f av = *(const v4f*)(A + (size_t)(m0 + row) * K + k0 + col);
                v4bf ab = { f2bf(av.x), f2bf(av.y), f2bf(av.z), f2bf(av.w) };
                *(v4bf*)(dst + row * LDT + col) = ab;
            }
        } else {
            // bf16 source: raw async copy, 16B per lane per issue
            #pragma unroll
            for (int i = 0; i < 4; ++i) {
                int c   = tid + i * 256;       // 1024 x 16B chunks
                int row = c >> 3;
                int seg = (c & 7) * 8;         // halves
                async_g2l_b128(dst + row * LDT + seg,
                               A + (size_t)(m0 + row) * K + k0 + seg);
            }
        }
    };
    auto compute = [&](const bf16_t* a, const bf16_t* b) {
        #pragma unroll
        for (int kf = 0; kf < 2; ++kf) {
            v16bf af[2];
            #pragma unroll
            for (int mt = 0; mt < 2; ++mt) {
                int row = wm * 32 + mt * 16 + lrow;
                int kb  = kf * 32 + lhi * 8;
                v8bf lo = *(const v8bf*)(a + row * LDT + kb);
                v8bf hi = *(const v8bf*)(a + row * LDT + kb + 16);
                af[mt] = cat8(lo, hi);
            }
            #pragma unroll
            for (int nt = 0; nt < 4; ++nt) {
                int col = wn * 64 + nt * 16 + lrow;
                int kb  = kf * 32 + lhi * 8;
                v8bf lo = *(const v8bf*)(b + col * LDT + kb);
                v8bf hi = *(const v8bf*)(b + col * LDT + kb + 16);
                v16bf bfrag = cat8(lo, hi);
                #pragma unroll
                for (int mt = 0; mt < 2; ++mt) {
                    acc[mt][nt] = __builtin_amdgcn_wmma_f32_16x16x32_bf16(
                        false, af[mt], false, bfrag, (short)0, acc[mt][nt],
                        false, false);
                }
            }
        }
    };

    // prologue: stage first tile
    stageW(bs[0], 0);
    stageA(as[0], 0);
    if constexpr (sizeof(TA) == 2) wait_async0();
    __syncthreads();

    int buf = 0;
    for (int k0 = 0; k0 < K; k0 += 64) {
        int nk = k0 + 64;
        if (nk < K) {                       // stage next tile into alt buffer
            stageW(bs[buf ^ 1], nk);
            stageA(as[buf ^ 1], nk);
        }
        compute(as[buf], bs[buf]);          // overlaps with staging loads
        if constexpr (sizeof(TA) == 2) wait_async0();
        __syncthreads();
        buf ^= 1;
    }

    // N, K compile-time: all row offsets fold into store immediates
    #pragma unroll
    for (int mt = 0; mt < 2; ++mt) {
        #pragma unroll
        for (int nt = 0; nt < 4; ++nt) {
            #pragma unroll
            for (int r = 0; r < 8; ++r) {
                size_t idx = (size_t)(m0 + wm * 32 + mt * 16 + lhi * 8 + r) * N
                           + n0 + wn * 64 + nt * 16 + lrow;
                if constexpr (sizeof(TO) == 4) C[idx] = acc[mt][nt][r];
                else                           C[idx] = f2bf(acc[mt][nt][r]);
            }
        }
    }
}

// ================================================================
// Causal flash attention over bf16 qkv. Block = (head, 128 query rows),
// 8 waves x 16 rows; 64-key chunks; ping-pong K/V buffers, K via async.
// ================================================================
#define LDK 136   // K-chunk stride (64 rows x 128 d)
#define LDV 72    // V^T stride     (128 d x 64 keys)
#define LDP 80    // per-wave P tile stride (16 rows x 64 keys)

__global__ __launch_bounds__(256)
void flash_attn_kernel(const bf16_t* __restrict__ qkv, bf16_t* __restrict__ out) {
    __shared__ bf16_t kbuf[2][64 * LDK];
    __shared__ bf16_t vt[2][128 * LDV];
    __shared__ bf16_t pbuf[8][16 * LDP];

    const int head = blockIdx.x;            // 0..15
    const int qb   = blockIdx.y;            // 0..31
    const int g    = head >> 2;
    const int qi   = head & 3;
    const int qoff = g * GROUP + qi * DHEAD;
    const int koff = g * GROUP + QPG * DHEAD;
    const int voff = g * GROUP + (QPG + 1) * DHEAD;

    const int tid  = threadIdx.x;
    const int lane = tid & 31;
    const int w    = tid >> 5;
    const int lrow = lane & 15;
    const int lhi  = lane >> 4;

    const int   qrow  = qb * 128 + w * 16;
    const float scale = 0.08838834764831845f;   // 1/sqrt(128)
    const float L2E   = 1.4426950408889634f;

    // Q fragments (A layout): raw bf16 b128 loads from global
    v16bf qfrag[4];
    {
        const bf16_t* qp = qkv + (size_t)(qrow + lrow) * TOTAL + qoff;
        #pragma unroll
        for (int f = 0; f < 4; ++f) {
            int kb = f * 32 + lhi * 8;
            v8bf lo = *(const v8bf*)(qp + kb);
            v8bf hi = *(const v8bf*)(qp + kb + 16);
            qfrag[f] = cat8(lo, hi);
        }
    }

    v8f o[8];
    #pragma unroll
    for (int nt = 0; nt < 8; ++nt) o[nt] = zero8();
    float m[8], l[8];
    #pragma unroll
    for (int r = 0; r < 8; ++r) { m[r] = -3.0e38f; l[r] = 0.0f; }

    const int kend = qb * 128 + 127;
    const int vrow = tid >> 2;          // 0..63
    const int vcb  = (tid & 3) * 32;    // 0,32,64,96

    auto load_kv = [&](int bi, int kb) {
        // K chunk (64x128, row major): async global->LDS
        #pragma unroll
        for (int i = 0; i < 4; ++i) {
            int c   = tid + i * 256;        // 1024 x 16B chunks
            int row = c >> 4;
            int seg = (c & 15) * 8;         // halves
            async_g2l_b128(kbuf[bi] + row * LDK + seg,
                           qkv + (size_t)(kb + row) * TOTAL + koff + seg);
        }
        // V chunk transposed into LDS (manual scatter)
        #pragma unroll
        for (int i = 0; i < 4; ++i) {
            int col0 = vcb + i * 8;
            v8bf vv = *(const v8bf*)(qkv + (size_t)(kb + vrow) * TOTAL + voff + col0);
            #pragma unroll
            for (int j = 0; j < 8; ++j)
                vt[bi][(col0 + j) * LDV + vrow] = vv[j];
        }
    };

    // prologue
    load_kv(0, 0);
    wait_async0();
    __syncthreads();

    int buf = 0;
    for (int kb = 0; kb <= kend; kb += 64) {
        int nkb = kb + 64;
        if (nkb <= kend) load_kv(buf ^ 1, nkb);   // prefetch next chunk

        if (kb <= qrow + 15) {   // wave-uniform causal block skip
            const bf16_t* kp_base = kbuf[buf];
            const bf16_t* vp_base = vt[buf];
            // ---- S = Q * K^T : 4 key tiles x 4 D-slices ----
            v8f s[4];
            #pragma unroll
            for (int kt = 0; kt < 4; ++kt) s[kt] = zero8();
            #pragma unroll
            for (int kt = 0; kt < 4; ++kt) {
                #pragma unroll
                for (int f = 0; f < 4; ++f) {
                    int kb2 = f * 32 + lhi * 8;
                    const bf16_t* kp = kp_base + (kt * 16 + lrow) * LDK + kb2;
                    v16bf kf = cat8(*(const v8bf*)kp, *(const v8bf*)(kp + 16));
                    s[kt] = __builtin_amdgcn_wmma_f32_16x16x32_bf16(
                        false, qfrag[f], false, kf, (short)0, s[kt], false, false);
                }
            }
            // ---- mask + scale + online softmax ----
            float rscale[8];
            #pragma unroll
            for (int r = 0; r < 8; ++r) {
                int row = qrow + r + lhi * 8;
                float sv[4];
                #pragma unroll
                for (int kt = 0; kt < 4; ++kt) {
                    float v = s[kt][r] * scale;
                    if (kb + kt * 16 + lrow > row) v = -3.0e38f;
                    sv[kt] = v;
                }
                float mx = fmaxf(fmaxf(sv[0], sv[1]), fmaxf(sv[2], sv[3]));
                mx = fmaxf(mx, __shfl_xor(mx, 1));
                mx = fmaxf(mx, __shfl_xor(mx, 2));
                mx = fmaxf(mx, __shfl_xor(mx, 4));
                mx = fmaxf(mx, __shfl_xor(mx, 8));
                float mnew = fmaxf(m[r], mx);
                float rs = 0.f;
                #pragma unroll
                for (int kt = 0; kt < 4; ++kt) {
                    float p = exp2f((sv[kt] - mnew) * L2E);
                    pbuf[w][(r + lhi * 8) * LDP + kt * 16 + lrow] = f2bf(p);
                    rs += p;
                }
                rs += __shfl_xor(rs, 1);
                rs += __shfl_xor(rs, 2);
                rs += __shfl_xor(rs, 4);
                rs += __shfl_xor(rs, 8);
                float sc = exp2f((m[r] - mnew) * L2E);
                rscale[r] = sc;
                l[r] = l[r] * sc + rs;
                m[r] = mnew;
            }
            // ---- O = diag(rscale)*O + P * V ----
            v16bf pf[2];
            #pragma unroll
            for (int pk = 0; pk < 2; ++pk) {
                int kb2 = pk * 32 + lhi * 8;
                v8bf lo = *(const v8bf*)(pbuf[w] + lrow * LDP + kb2);
                v8bf hi = *(const v8bf*)(pbuf[w] + lrow * LDP + kb2 + 16);
                pf[pk] = cat8(lo, hi);
            }
            #pragma unroll
            for (int nt = 0; nt < 8; ++nt) {
                v8f oc = o[nt];
                #pragma unroll
                for (int r = 0; r < 8; ++r) oc[r] *= rscale[r];
                #pragma unroll
                for (int pk = 0; pk < 2; ++pk) {
                    int dcol = nt * 16 + lrow;
                    int kb2  = pk * 32 + lhi * 8;
                    v8bf lo = *(const v8bf*)(vp_base + dcol * LDV + kb2);
                    v8bf hi = *(const v8bf*)(vp_base + dcol * LDV + kb2 + 16);
                    oc = __builtin_amdgcn_wmma_f32_16x16x32_bf16(
                        false, pf[pk], false, cat8(lo, hi), (short)0, oc,
                        false, false);
                }
                o[nt] = oc;
            }
        }
        wait_async0();        // next-chunk asyncs complete before barrier
        __syncthreads();
        buf ^= 1;
    }

    // normalize and store bf16 [s, head*128 + d]
    #pragma unroll
    for (int r = 0; r < 8; ++r) l[r] = 1.0f / l[r];
    #pragma unroll
    for (int nt = 0; nt < 8; ++nt) {
        #pragma unroll
        for (int r = 0; r < 8; ++r) {
            size_t idx = (size_t)(qrow + r + lhi * 8) * HID
                       + head * DHEAD + nt * 16 + lrow;
            out[idx] = f2bf(o[nt][r] * l[r]);
        }
    }
}

// ================================================================
extern "C" void kernel_launch(void* const* d_in, const int* in_sizes, int n_in,
                              void* d_out, int out_size, void* d_ws, size_t ws_size,
                              hipStream_t stream) {
    const float* x  = (const float*)d_in[0];   // [1,4096,2048]
    const float* wq = (const float*)d_in[1];   // [3072,2048]
    const float* wp = (const float*)d_in[2];   // [2048,2048]
    float* outp = (float*)d_out;               // [1,4096,2048] fp32

    bf16_t* qkv  = (bf16_t*)d_ws;                      // 4096*3072 bf16
    bf16_t* attn = qkv + (size_t)S_LEN * TOTAL;        // 4096*2048 bf16

    dim3 blk(256);
    gemm_xwT<TOTAL, HID, float, bf16_t>
        <<<dim3(TOTAL / 128, S_LEN / 128), blk, 0, stream>>>(x, wq, qkv);
    flash_attn_kernel<<<dim3(NH, S_LEN / 128), blk, 0, stream>>>(qkv, attn);
    gemm_xwT<HID, HID, bf16_t, float>
        <<<dim3(HID / 128, S_LEN / 128), blk, 0, stream>>>(attn, wp, outp);
}
